// CrfLossLayer_10256381903472
// MI455X (gfx1250) — matrix operand order repaired and checked by reading, not compiled, not run on hardware
//
#include <hip/hip_runtime.h>

typedef __attribute__((ext_vector_type(16))) _Float16 v16h;
typedef __attribute__((ext_vector_type(8)))  _Float16 v8h;
typedef __attribute__((ext_vector_type(8)))  float    v8f;
typedef __attribute__((ext_vector_type(4)))  float    v4f;

#define B_ 2048
#define T_ 120
#define K_ 128

#define LOG2E_F 1.44269504088896340736f
#define LN2_F   0.69314718055994530942f

// ---------------- kernel 1: Et[j][k] = (f16) exp(trans[k][j]) ----------------
__global__ void crf_prep_Et(const float* __restrict__ trans,
                            _Float16* __restrict__ Et) {
    int idx = blockIdx.x * blockDim.x + threadIdx.x;   // 0..16383
    int j = idx >> 7;
    int k = idx & 127;
    Et[j * K_ + k] = (_Float16)__builtin_amdgcn_exp2f(trans[k * K_ + j] * LOG2E_F);
}

// ---------------- kernel 2: gold-path score per batch ----------------
__global__ void crf_score(const float* __restrict__ emis,
                          const int*   __restrict__ tags,
                          const int*   __restrict__ mask,
                          const float* __restrict__ trans,
                          float*       __restrict__ score) {
    int b = blockIdx.x * blockDim.x + threadIdx.x;
    if (b >= B_) return;
    const int*   tg = tags + b * T_;
    const int*   mk = mask + b * T_;
    const float* em = emis + (size_t)b * T_ * K_;
    int   prev  = tg[0];
    float mprev = (float)mk[0];
    float s = em[prev] * mprev;
    for (int t = 1; t < T_; ++t) {
        int   cur = tg[t];
        float mt  = (float)mk[t];
        s += em[(size_t)t * K_ + cur] * mt;
        s += trans[prev * K_ + cur] * (mprev * mt);
        prev = cur; mprev = mt;
    }
    score[b] = s;
}

// ---------------- kernel 3: WMMA forward recursion ----------------
// grid = 128 workgroups (16 batch rows each), block = 256 threads = 8 waves.
// alpha lives in REGISTERS in the WMMA C layout: lane (half,n) of wave w owns
// rows m = r + 8*half (r=0..7) at column j = 16*w + n. Per step:
//   - tile max: 32-lane butterfly + 8-float LDS line (stability shift only)
//   - ea = exp2((alpha-mx)*log2e) scattered f16 stores into eaS (A layout)
//   - 8x ds_load_b128 -> 4x v_wmma_f32_16x16x32_f16 (K=128)
//   - alpha' = ln2*log2(sum) + mx + emis (register double-buffered NT loads)
__global__ __launch_bounds__(256) void crf_forward(
    const float*    __restrict__ emis,
    const int*      __restrict__ mask,
    const _Float16* __restrict__ Et,
    const float*    __restrict__ score,
    float*          __restrict__ ll)
{
    __shared__ _Float16 eaS[16][144];   // exp(alpha - mx), 16B-aligned rows
    __shared__ float    pmaxS[8];       // per-wave tile-max partials
    __shared__ float    redS[8][16];    // final per-row reduction scratch
    __shared__ float    rowredS[16];

    const int tid  = threadIdx.x;
    const int wave = tid >> 5;
    const int lane = tid & 31;
    const int half = lane >> 4;     // 0/1: M offset (rows 0-7 vs 8-15)
    const int n    = lane & 15;     // column within wave tile / A-matrix row
    const int jn   = wave * 16;
    const int j    = jn + n;        // owned column
    const int b0   = blockIdx.x * 16;

    // ---- loop-invariant B fragments: Et col tile, K=0..127 ----
    v16h bfrag[4];
#pragma unroll
    for (int kb = 0; kb < 4; ++kb) {
        const _Float16* p = Et + (size_t)(jn + n) * K_ + kb * 32 + half * 16;
        bfrag[kb] = *(const v16h*)p;
    }

    // per-lane bases for its 8 rows (m = r + 8*half), column j
    const float* embase = emis + (size_t)(b0 + half * 8) * T_ * K_ + j;
    const int*   mkbase = mask + (b0 + half * 8) * T_;
    const size_t rstr   = (size_t)T_ * K_;   // row stride in emissions

    // ---- alpha(0) = emissions[:,0,j]; preload t=1 emis+mask ----
    float av[8], en[8];
    int   mkn[8];
#pragma unroll
    for (int r = 0; r < 8; ++r) {
        av[r]  = embase[r * rstr];
        en[r]  = __builtin_nontemporal_load(embase + r * rstr + K_);
        mkn[r] = mkbase[r * T_ + 1];
    }

    for (int t = 1; t < T_; ++t) {
        float ec[8];
        int   mkc[8];
#pragma unroll
        for (int r = 0; r < 8; ++r) { ec[r] = en[r]; mkc[r] = mkn[r]; }

        // ---- tile max (stability shift): wave butterfly + 8-float LDS ----
        float mw = av[0];
#pragma unroll
        for (int r = 1; r < 8; ++r) mw = fmaxf(mw, av[r]);
#pragma unroll
        for (int off = 1; off < 32; off <<= 1)
            mw = fmaxf(mw, __shfl_xor(mw, off, 32));
        if (lane == 0) pmaxS[wave] = mw;
        __syncthreads();
        v4f p0 = *(const v4f*)&pmaxS[0];
        v4f p1 = *(const v4f*)&pmaxS[4];
        float mx = fmaxf(fmaxf(fmaxf(p0.x, p0.y), fmaxf(p0.z, p0.w)),
                         fmaxf(fmaxf(p1.x, p1.y), fmaxf(p1.z, p1.w)));

        // ---- ea tile into LDS (A-matrix layout: eaS[m][k]) ----
#pragma unroll
        for (int r = 0; r < 8; ++r) {
            int m = r + half * 8;
            eaS[m][j] = (_Float16)__builtin_amdgcn_exp2f((av[r] - mx) * LOG2E_F);
        }

        // ---- issue next-step (t+1) emission/mask loads (hidden ~1 iter) ----
        if (t + 1 < T_) {
#pragma unroll
            for (int r = 0; r < 8; ++r) {
                en[r]  = __builtin_nontemporal_load(embase + r * rstr + (size_t)(t + 1) * K_);
                mkn[r] = mkbase[r * T_ + (t + 1)];
            }
        }
        __builtin_prefetch(embase + (size_t)((t + 2 < T_) ? t + 2 : T_ - 1) * K_, 0, 3);
        __syncthreads();

        // ---- phase 2: sums = ea(16x128 f16) x E(128x16 f16), f32 acc ----
        v8h alo[4], ahi[4];
#pragma unroll
        for (int kb = 0; kb < 4; ++kb) {
            alo[kb] = *(const v8h*)&eaS[n][kb * 32 + half * 16];
            ahi[kb] = *(const v8h*)&eaS[n][kb * 32 + half * 16 + 8];
        }
        v8f acc = {};
#pragma unroll
        for (int kb = 0; kb < 4; ++kb) {
            v16h afrag = __builtin_shufflevector(alo[kb], ahi[kb],
                0, 1, 2, 3, 4, 5, 6, 7, 8, 9, 10, 11, 12, 13, 14, 15);
            acc = __builtin_amdgcn_wmma_f32_16x16x32_f16(
                false, afrag, false, bfrag[kb], (short)0, acc, false, false);
        }

        // ---- epilogue: alpha' in registers, masked ----
#pragma unroll
        for (int r = 0; r < 8; ++r) {
            float nv = fmaf(LN2_F, __builtin_amdgcn_logf(acc[r]), mx + ec[r]);
            av[r] = (mkc[r] > 0) ? nv : av[r];
        }
    }

    // ---- final per-row logsumexp + ll = score - lognorm ----
    {
        // pass 1: exact per-row max (16-lane butterfly, then across 8 waves)
        float bm[8];
#pragma unroll
        for (int r = 0; r < 8; ++r) {
            bm[r] = av[r];
#pragma unroll
            for (int off = 1; off < 16; off <<= 1)
                bm[r] = fmaxf(bm[r], __shfl_xor(bm[r], off, 16));
        }
        if (n == 0) {
#pragma unroll
            for (int r = 0; r < 8; ++r) redS[wave][r + half * 8] = bm[r];
        }
        __syncthreads();
        if (tid < 16) {
            float v = redS[0][tid];
#pragma unroll
            for (int w = 1; w < 8; ++w) v = fmaxf(v, redS[w][tid]);
            rowredS[tid] = v;
        }
        __syncthreads();
        // pass 2: per-row sum of exp
        float sm[8];
#pragma unroll
        for (int r = 0; r < 8; ++r) {
            int m = r + half * 8;
            sm[r] = __builtin_amdgcn_exp2f((av[r] - rowredS[m]) * LOG2E_F);
#pragma unroll
            for (int off = 1; off < 16; off <<= 1)
                sm[r] += __shfl_xor(sm[r], off, 16);
        }
        if (n == 0) {
#pragma unroll
            for (int r = 0; r < 8; ++r) redS[wave][r + half * 8] = sm[r];
        }
        __syncthreads();
        if (tid < 16) {
            float s = redS[0][tid];
#pragma unroll
            for (int w = 1; w < 8; ++w) s += redS[w][tid];
            ll[b0 + tid] = score[b0 + tid]
                         - (LN2_F * __builtin_amdgcn_logf(s) + rowredS[tid]);
        }
    }
}

// ---------------- kernel 4: deterministic mean reduction ----------------
__global__ void crf_reduce(const float* __restrict__ ll, float* __restrict__ out) {
    __shared__ float buf[256];
    int tid = threadIdx.x;
    float s = 0.f;
    for (int i = tid; i < B_; i += 256) s += ll[i];   // fixed order per thread
    buf[tid] = s;
    __syncthreads();
    for (int st = 128; st > 0; st >>= 1) {
        if (tid < st) buf[tid] += buf[tid + st];
        __syncthreads();
    }
    if (tid == 0) out[0] = -buf[0] / (float)B_;
}

extern "C" void kernel_launch(void* const* d_in, const int* in_sizes, int n_in,
                              void* d_out, int out_size, void* d_ws, size_t ws_size,
                              hipStream_t stream) {
    const float* emis  = (const float*)d_in[0];   // (B,T,K) f32
    const int*   tags  = (const int*)d_in[1];     // (B,T)
    const int*   mask  = (const int*)d_in[2];     // (B,T)
    const float* trans = (const float*)d_in[3];   // (K,K) f32

    _Float16* Et    = (_Float16*)d_ws;                          // 32 KB
    float*    score = (float*)((char*)d_ws + 32768);            //  8 KB
    float*    ll    = (float*)((char*)d_ws + 32768 + 8192);     //  8 KB

    crf_prep_Et<<<64, 256, 0, stream>>>(trans, Et);
    crf_score<<<8, 256, 0, stream>>>(emis, tags, mask, trans, score);
    crf_forward<<<B_ / 16, 256, 0, stream>>>(emis, mask, Et, score, ll);
    crf_reduce<<<1, 256, 0, stream>>>(ll, (float*)d_out);
}
